// TrainablePQCQCNN_52338471469482
// MI455X (gfx1250) — matrix operand order repaired and checked by reading, not compiled, not run on hardware
//
#include <hip/hip_runtime.h>
#include <hip/hip_bf16.h>

// ---------------------------------------------------------------------------
// MI455X (gfx1250): wave32, WMMA 16x16x32 f16 GEMMs, async-to-LDS staging.
// Memory-bound pipeline (~50MB @ 23.3TB/s). Quantum sim: one 6-qubit state
// per wave (2 complex amps per lane); shared gate matrices precomputed once.
// ---------------------------------------------------------------------------

typedef __attribute__((ext_vector_type(16))) _Float16 v16h;
typedef __attribute__((ext_vector_type(8)))  _Float16 h8;
typedef __attribute__((ext_vector_type(4)))  _Float16 h4;
typedef __attribute__((ext_vector_type(8)))  float    v8f;
typedef __attribute__((ext_vector_type(4)))  int      v4i;

union V16H { v16h v; _Float16 h[16]; };

#define NQ 6
#define FEAT 512
#define HID 32
#define CCH 128
#define NOUT 240

#if defined(__has_builtin)
#if __has_builtin(__builtin_amdgcn_global_load_async_to_lds_b128) && \
    __has_builtin(__builtin_amdgcn_s_wait_asynccnt)
#define USE_ASYNC_COPY 1
#endif
#endif
#ifndef USE_ASYNC_COPY
#define USE_ASYNC_COPY 0
#endif

// ------------------------------ Kernel 1 -----------------------------------
// hidden = tanh(x @ W1^T + b1)  [B,32]  via v_wmma_f32_16x16x32_f16
// angles = hidden @ W2^T + b2   [B,6]
__global__ __launch_bounds__(256)
void enc_kernel(const float* __restrict__ x,
                const float* __restrict__ w1,
                const float* __restrict__ bias1,
                const float* __restrict__ w2,
                const float* __restrict__ bias2,
                float* __restrict__ angles)
{
    // Per-wave staged x tile: 16 rows x 64 K-values f32 (row padded to 68 to
    // break the 64-bank alignment); wave-private => no block barrier needed.
    __shared__ __align__(16) float xs[8][16][68];
    __shared__ float hbuf[8][16][33];

    const int tid  = threadIdx.x;
    const int wv   = tid >> 5;
    const int lane = tid & 31;
    const int hi   = lane >> 4;
    const int lln  = lane & 15;
    const int m0   = (blockIdx.x * 8 + wv) * 16;

    v8f acc0 = {}; v8f acc1 = {};
    const float* b0r = w1 + (lln)      * FEAT;   // B tile0 column = W1 row lln
    const float* b1r = w1 + (16 + lln) * FEAT;   // B tile1 column = W1 row 16+lln

    for (int kc = 0; kc < FEAT; kc += 64) {
        if (kc + 64 < FEAT) __builtin_prefetch(b0r + kc + 64, 0, 3); // global_prefetch_b8
        // ---- stage 16x64 f32 tile of x into LDS (coalesced: 16B/lane, rows of 256B)
        #pragma unroll
        for (int i = 0; i < 8; ++i) {
            const int lin = i * 32 + lane;       // 0..255 16B-chunks
            const int row = lin >> 4;            // 0..15
            const int c16 = lin & 15;            // chunk within row
            const float* gsrc = x + (size_t)(m0 + row) * FEAT + kc + c16 * 4;
            float* ldst = &xs[wv][row][c16 * 4];
#if USE_ASYNC_COPY
            __builtin_amdgcn_global_load_async_to_lds_b128(
                (__attribute__((address_space(1))) v4i*)gsrc,
                (__attribute__((address_space(3))) v4i*)ldst, 0, 0);
#else
            *(float4*)ldst = *(const float4*)gsrc;
#endif
        }
#if USE_ASYNC_COPY
        __builtin_amdgcn_s_wait_asynccnt(0);
#endif
        // ---- two WMMA K-steps per staged chunk
        #pragma unroll
        for (int kk = 0; kk < 64; kk += 32) {
            const int k = kc + kk;
            // A fragment: per-lane 16 contiguous K at k+hi*16, from LDS
            V16H a;
            const float* ar = &xs[wv][lln][kk];
            #pragma unroll
            for (int j = 0; j < 8; ++j) {
                a.h[j]     = (_Float16)ar[hi * 16 + j];
                a.h[8 + j] = (_Float16)ar[hi * 16 + 8 + j];
            }
            // B fragments: per-lane 16 contiguous K at k+hi*16 of one column
            V16H b0, b1v;
            const float4* p0 = (const float4*)(b0r + k + hi * 16);
            const float4* p1 = (const float4*)(b1r + k + hi * 16);
            #pragma unroll
            for (int j = 0; j < 4; ++j) {
                float4 t0 = p0[j], t1 = p1[j];
                b0.h[4*j+0]=(_Float16)t0.x; b0.h[4*j+1]=(_Float16)t0.y;
                b0.h[4*j+2]=(_Float16)t0.z; b0.h[4*j+3]=(_Float16)t0.w;
                b1v.h[4*j+0]=(_Float16)t1.x; b1v.h[4*j+1]=(_Float16)t1.y;
                b1v.h[4*j+2]=(_Float16)t1.z; b1v.h[4*j+3]=(_Float16)t1.w;
            }
            acc0 = __builtin_amdgcn_wmma_f32_16x16x32_f16(false, a.v, false, b0.v,  (short)0, acc0, false, false);
            acc1 = __builtin_amdgcn_wmma_f32_16x16x32_f16(false, a.v, false, b1v.v, (short)0, acc1, false, false);
        }
    }

    // C layout: VGPR r -> row m0 + r + hi*8, col lln (acc0) / 16+lln (acc1)
    #pragma unroll
    for (int r = 0; r < 8; ++r) {
        hbuf[wv][r + hi * 8][lln]      = tanhf(acc0[r] + bias1[lln]);
        hbuf[wv][r + hi * 8][16 + lln] = tanhf(acc1[r] + bias1[16 + lln]);
    }
    __syncthreads();

    if (lane < 16) {
        const int row = m0 + lane;
        float h[HID];
        #pragma unroll
        for (int n = 0; n < HID; ++n) h[n] = hbuf[wv][lane][n];
        #pragma unroll
        for (int q = 0; q < NQ; ++q) {
            float s = bias2[q];
            #pragma unroll
            for (int n = 0; n < HID; ++n) s += h[n] * w2[q * HID + n];
            angles[row * NQ + q] = s;
        }
    }
}

// ----------------------- complex / gate helpers ----------------------------
struct cpx { float x, y; };
__device__ __forceinline__ cpx cmul(cpx a, cpx b) { return {a.x*b.x - a.y*b.y, a.x*b.y + a.y*b.x}; }
__device__ __forceinline__ cpx cadd(cpx a, cpx b) { return {a.x + b.x, a.y + b.y}; }
__device__ __forceinline__ cpx cshx(cpx a, int m) { return { __shfl_xor(a.x, m, 32), __shfl_xor(a.y, m, 32) }; }
struct C2 { cpx u00, u01, u10, u11; };

__device__ __forceinline__ C2 u3m(float t, float p, float l) {
    float st, ct; __sincosf(0.5f * t, &st, &ct);
    float sp, cp; __sincosf(p, &sp, &cp);
    float sl, cl; __sincosf(l, &sl, &cl);
    float spl, cpl; __sincosf(p + l, &spl, &cpl);
    C2 U;
    U.u00 = { ct, 0.f };
    U.u01 = { -cl * st, -sl * st };
    U.u10 = {  cp * st,  sp * st };
    U.u11 = { cpl * ct, spl * ct };
    return U;
}
__device__ __forceinline__ C2 rym(float t) {
    float s, c; __sincosf(0.5f * t, &s, &c);
    return { {c,0.f}, {-s,0.f}, {s,0.f}, {c,0.f} };
}
__device__ __forceinline__ C2 rzm(float t) {
    float s, c; __sincosf(0.5f * t, &s, &c);
    return { {c,-s}, {0.f,0.f}, {0.f,0.f}, {c,s} };
}
__device__ __forceinline__ C2 ldgate(const float* __restrict__ g) {
    return C2{ {g[0],g[1]}, {g[2],g[3]}, {g[4],g[5]}, {g[6],g[7]} };
}

// qubit q <-> state-index bit (5-q); bit5 = register slot, bits4..0 = lane
__device__ __forceinline__ void apply1(cpx& a0, cpx& a1, int lane, int q, const C2& U) {
    if (q == 0) {               // slot-bit qubit: fully in-lane
        cpx n0 = cadd(cmul(U.u00, a0), cmul(U.u01, a1));
        cpx n1 = cadd(cmul(U.u10, a0), cmul(U.u11, a1));
        a0 = n0; a1 = n1;
    } else {                    // lane-bit qubit: pair via shfl_xor
        const int bit = 5 - q, m = 1 << bit;
        const bool up = (lane >> bit) & 1;
        cpx p0 = cshx(a0, m), p1 = cshx(a1, m);
        cpx n0 = up ? cadd(cmul(U.u10, p0), cmul(U.u11, a0))
                    : cadd(cmul(U.u00, a0), cmul(U.u01, p0));
        cpx n1 = up ? cadd(cmul(U.u10, p1), cmul(U.u11, a1))
                    : cadd(cmul(U.u00, a1), cmul(U.u01, p1));
        a0 = n0; a1 = n1;
    }
}

__device__ __forceinline__ void cnotg(cpx& a0, cpx& a1, int lane, int c, int t) {
    if (t == 0) {                               // target = slot bit
        if ((lane >> (5 - c)) & 1) { cpx tmp = a0; a0 = a1; a1 = tmp; }
    } else if (c == 0) {                        // control = slot bit: permute slot-1 amps
        a1 = cshx(a1, 1 << (5 - t));
    } else {                                    // both on lane bits
        const int m = 1 << (5 - t);
        cpx p0 = cshx(a0, m), p1 = cshx(a1, m);
        if ((lane >> (5 - c)) & 1) { a0 = p0; a1 = p1; }
    }
}

// --------------------------- Gate precompute -------------------------------
// 18 c9 blocks x 7 shared gates -> 126 complex 2x2 matrices (8 floats each).
__global__ __launch_bounds__(128)
void gate_prep_kernel(const float* __restrict__ qw, float* __restrict__ gates)
{
    const int t = threadIdx.x;
    if (t >= 126) return;
    const int blk = t / 7, g = t % 7;
    const float* p = qw + blk * 15;
    C2 U;
    switch (g) {
        case 0:  U = u3m(p[0],  p[1],  p[2]);  break;
        case 1:  U = u3m(p[3],  p[4],  p[5]);  break;
        case 2:  U = rym(p[6]);                break;
        case 3:  U = rzm(p[7]);                break;
        case 4:  U = rym(p[8]);                break;
        case 5:  U = u3m(p[9],  p[10], p[11]); break;
        default: U = u3m(p[12], p[13], p[14]); break;
    }
    float* o = gates + t * 8;
    o[0]=U.u00.x; o[1]=U.u00.y; o[2]=U.u01.x; o[3]=U.u01.y;
    o[4]=U.u10.x; o[5]=U.u10.y; o[6]=U.u11.x; o[7]=U.u11.y;
}

// ------------------------------ Kernel 2 -----------------------------------
// wave-per-sample 6-qubit statevector sim + <Z_q> + conv1d/relu/mean-pool.
__global__ __launch_bounds__(256)
void qsim_kernel(const float* __restrict__ angles,
                 const float* __restrict__ gates,   // [18][7][8]
                 const float* __restrict__ conv_w,  // [128][3]
                 const float* __restrict__ conv_b,  // [128]
                 _Float16* __restrict__ pooled)     // [B][128]
{
    const int lane = threadIdx.x & 31;
    const int wv   = threadIdx.x >> 5;
    const int b    = blockIdx.x * 8 + wv;

    cpx a0 = {0.f, 0.f}, a1 = {0.f, 0.f};
    if (lane == 0) a0.x = 1.0f;                 // |000000>

    // per-sample RY-encoding matrices: compute sincos once (reused each layer)
    float ec[NQ], es[NQ];
    #pragma unroll
    for (int q = 0; q < NQ; ++q) {
        const float a = angles[b * NQ + q];
        __sincosf(0.5f * a, &es[q], &ec[q]);
    }

    const int P0[6] = {0, 2, 4, 1, 3, 5};
    const int P1[6] = {1, 3, 5, 2, 4, 0};

    for (int layer = 0; layer < 3; ++layer) {
        #pragma unroll
        for (int q = 0; q < NQ; ++q) {
            C2 R{ {ec[q],0.f}, {-es[q],0.f}, {es[q],0.f}, {ec[q],0.f} };
            apply1(a0, a1, lane, q, R);
        }
        #pragma unroll
        for (int bi = 0; bi < 6; ++bi) {
            const float* gb = gates + (layer * 6 + bi) * 7 * 8;   // wave-uniform
            const int w0 = P0[bi], w1 = P1[bi];
            apply1(a0, a1, lane, w0, ldgate(gb + 0));   // u3
            apply1(a0, a1, lane, w1, ldgate(gb + 8));   // u3
            cnotg (a0, a1, lane, w0, w1);
            apply1(a0, a1, lane, w0, ldgate(gb + 16));  // ry
            apply1(a0, a1, lane, w1, ldgate(gb + 24));  // rz
            cnotg (a0, a1, lane, w1, w0);
            apply1(a0, a1, lane, w0, ldgate(gb + 32));  // ry
            cnotg (a0, a1, lane, w0, w1);
            apply1(a0, a1, lane, w0, ldgate(gb + 40));  // u3
            apply1(a0, a1, lane, w1, ldgate(gb + 48));  // u3
        }
    }

    // <Z_q>: q=0 keyed on slot bit, q>=1 on lane bits; butterfly wave reduce
    const float p0 = a0.x * a0.x + a0.y * a0.y;
    const float p1 = a1.x * a1.x + a1.y * a1.y;
    float z[NQ];
    z[0] = p0 - p1;
    #pragma unroll
    for (int q = 1; q < NQ; ++q) {
        const float sg = ((lane >> (5 - q)) & 1) ? -1.f : 1.f;
        z[q] = sg * (p0 + p1);
    }
    #pragma unroll
    for (int m = 1; m < 32; m <<= 1) {
        #pragma unroll
        for (int q = 0; q < NQ; ++q) z[q] += __shfl_xor(z[q], m, 32);
    }

    // conv1d(1->128,k=3,pad=1) + relu + mean over 6 positions; 4 channels/lane
    float zp[NQ + 2];
    zp[0] = 0.f; zp[NQ + 1] = 0.f;
    #pragma unroll
    for (int q = 0; q < NQ; ++q) zp[q + 1] = z[q];

    const int ch0 = lane * 4;
    h4 outv;
    #pragma unroll
    for (int i = 0; i < 4; ++i) {
        const int ch = ch0 + i;
        const float k0 = conv_w[ch * 3 + 0], k1 = conv_w[ch * 3 + 1], k2 = conv_w[ch * 3 + 2];
        const float cb = conv_b[ch];
        float s = 0.f;
        #pragma unroll
        for (int l = 0; l < NQ; ++l) {
            float v = fmaf(k0, zp[l], fmaf(k1, zp[l + 1], fmaf(k2, zp[l + 2], cb)));
            s += fmaxf(v, 0.f);
        }
        outv[i] = (_Float16)(s * (1.0f / 6.0f));
    }
    *(h4*)(pooled + b * CCH + ch0) = outv;
}

// ------------------------------ Kernel 3 -----------------------------------
// logits = pooled @ head_w^T + head_b  : [16384,128] x [128,240]
__global__ __launch_bounds__(256)
void head_kernel(const _Float16* __restrict__ pooled,
                 const float* __restrict__ head_w,   // [240][128]
                 const float* __restrict__ head_b,   // [240]
                 float* __restrict__ out)            // [B][240]
{
    const int tid  = threadIdx.x;
    const int wv   = tid >> 5;
    const int lane = tid & 31;
    const int hi   = lane >> 4;
    const int lln  = lane & 15;
    const int wave_id = blockIdx.x * 8 + wv;
    const int mt = wave_id / (NOUT / 16);
    const int nt = wave_id % (NOUT / 16);
    const int m0 = mt * 16, n0 = nt * 16;

    v8f acc = {};
    const _Float16* arow = pooled + (m0 + lln) * CCH;
    const float*    brow = head_w + (n0 + lln) * CCH;

    #pragma unroll
    for (int k = 0; k < CCH; k += 32) {
        V16H a;
        *(h8*)(&a.h[0]) = *(const h8*)(arow + k + hi * 16);
        *(h8*)(&a.h[8]) = *(const h8*)(arow + k + hi * 16 + 8);
        V16H bf;
        const float4* bp = (const float4*)(brow + k + hi * 16);
        #pragma unroll
        for (int j = 0; j < 4; ++j) {
            float4 t = bp[j];
            bf.h[4*j+0]=(_Float16)t.x; bf.h[4*j+1]=(_Float16)t.y;
            bf.h[4*j+2]=(_Float16)t.z; bf.h[4*j+3]=(_Float16)t.w;
        }
        acc = __builtin_amdgcn_wmma_f32_16x16x32_f16(false, a.v, false, bf.v, (short)0, acc, false, false);
    }

    const float bias = head_b[n0 + lln];
    #pragma unroll
    for (int r = 0; r < 8; ++r) {
        out[(m0 + r + hi * 8) * NOUT + n0 + lln] = acc[r] + bias;
    }
}

// ------------------------------ Launch -------------------------------------
extern "C" void kernel_launch(void* const* d_in, const int* in_sizes, int n_in,
                              void* d_out, int out_size, void* d_ws, size_t ws_size,
                              hipStream_t stream) {
    const float* x       = (const float*)d_in[0];
    const float* enc_w1  = (const float*)d_in[1];
    const float* enc_b1  = (const float*)d_in[2];
    const float* enc_w2  = (const float*)d_in[3];
    const float* enc_b2  = (const float*)d_in[4];
    const float* qw      = (const float*)d_in[5];
    const float* conv_w  = (const float*)d_in[6];
    const float* conv_b  = (const float*)d_in[7];
    const float* head_w  = (const float*)d_in[8];
    const float* head_b  = (const float*)d_in[9];
    float* out = (float*)d_out;

    const int B = in_sizes[0] / FEAT;   // 16384

    char* ws = (char*)d_ws;
    float*    ws_angles = (float*)ws;                                    // B*6 f32
    _Float16* ws_pooled = (_Float16*)(ws + (size_t)B * NQ * 4);          // B*128 f16
    float*    ws_gates  = (float*)(ws + (size_t)B * NQ * 4
                                      + (size_t)B * CCH * 2);            // 126*8 f32

    // shared-gate matrices (tiny, independent of x)
    gate_prep_kernel<<<1, 128, 0, stream>>>(qw, ws_gates);
    // encoder: 8 waves/block, 16 rows/wave -> 128 rows/block
    enc_kernel<<<B / 128, 256, 0, stream>>>(x, enc_w1, enc_b1, enc_w2, enc_b2, ws_angles);
    // quantum sim: one sample per wave, 8 samples/block
    qsim_kernel<<<B / 8, 256, 0, stream>>>(ws_angles, ws_gates, conv_w, conv_b, ws_pooled);
    // head GEMM: one 16x16 tile per wave
    head_kernel<<<(B / 16) * (NOUT / 16) / 8, 256, 0, stream>>>(ws_pooled, head_w, head_b, out);
}
